// MultiScaleGRU_2542620639312
// MI455X (gfx1250) — matrix operand order repaired and checked by reading, not compiled, not run on hardware
//
#include <hip/hip_runtime.h>

// MultiScaleGRU for MI455X (gfx1250, wave32, WMMA + async-to-LDS).
// - All big GEMMs: v_wmma_f32_16x16x32_bf16 (f32 accum), bf16 weights/activations.
//   Tiles staged with GLOBAL_LOAD_ASYNC_TO_LDS_B128 (ASYNCcnt-tracked DMA,
//   double-buffered LDS, s_wait_asynccnt fences) -> load/compute overlap without
//   VGPR staging. Partial-N tiles (classifier N=10) fall back to a guarded
//   synchronous loader.
// - GRU recurrence: 1 workgroup/channel persistent scan; whh r,z halves in
//   262KB LDS (CDNA5 320KB LDS/WG), n-row in VGPRs; 1 barrier per step.
// - ig = (x/s)@wih^T + b folded into GEMM epilogue as column scale exp(-clip(log_s)).

typedef unsigned short u16;
typedef unsigned int   u32;

typedef __attribute__((ext_vector_type(16))) __bf16 bf16x16;
typedef __attribute__((ext_vector_type(8)))  float  f32x8;

#define T_STEPS 8192
#define NINP    128
#define NSTATE  1024
#define NCH     4
#define NLAYER  3
#define NCLASS  10
#define GH      256
#define IGW     3072   // NCH * 3 * GH

// ---------------- helpers ----------------
__device__ __forceinline__ float bf_lo(u32 w){ return __uint_as_float(w << 16); }
__device__ __forceinline__ float bf_hi(u32 w){ return __uint_as_float(w & 0xffff0000u); }
__device__ __forceinline__ u16 f2bf(float f){
  u32 x = __float_as_uint(f);
  u32 r = ((x >> 16) & 1u) + 0x7fffu;   // round-to-nearest-even
  return (u16)((x + r) >> 16);
}

__global__ void cvt_f32_bf16(const float* __restrict__ in, u16* __restrict__ out, long n){
  long i  = (long)blockIdx.x * blockDim.x + threadIdx.x;
  long st = (long)gridDim.x * blockDim.x;
  for (; i < n; i += st) out[i] = f2bf(in[i]);
}

// Async DMA: global (saddr + 32-bit voffset) -> LDS (byte offset in VGPR).
// Tracked by ASYNCcnt; fence with s_wait_asynccnt.
__device__ __forceinline__ void async_ld_b128(u32 lds_byte_off, u32 gbyte_off, const void* sbase){
  asm volatile("global_load_async_to_lds_b128 %0, %1, %2"
               :: "v"(lds_byte_off), "v"(gbyte_off), "s"(sbase) : "memory");
}
__device__ __forceinline__ void wait_async_le4(){
  asm volatile("s_wait_asynccnt 4" ::: "memory");
}
__device__ __forceinline__ void wait_async_0(){
  asm volatile("s_wait_asynccnt 0" ::: "memory");
}

// ---------------- WMMA GEMM: C[M,N] = A[M,K] * B[N,K]^T (+epilogue) ----------------
#define BM 128
#define BN 128
#define BK 32
#define LDT 40   // padded LDS row stride (halves): conflict-free b128 frag reads

__device__ __forceinline__ bf16x16 frag_from(const u16* p0, const u16* p1){
  union { uint4 q[2]; bf16x16 v; } u;
  u.q[0] = *(const uint4*)p0;
  u.q[1] = *(const uint4*)p1;
  return u.v;
}

__global__ __launch_bounds__(256)
void gemm_bf16_wmma(const u16* __restrict__ A, const u16* __restrict__ Bw,
                    const float* __restrict__ bias, const float* __restrict__ logs,
                    const float* __restrict__ res,
                    float* __restrict__ outF, u16* __restrict__ outB,
                    int M, int N, int K, int relu)
{
  __shared__ u16 As[2][BM * LDT];
  __shared__ u16 Bs[2][BN * LDT];

  const int tid   = threadIdx.x;
  const int lane  = tid & 31;
  const int wave  = tid >> 5;
  const int waveM = (wave & 3) * 32;   // 4 waves along M
  const int waveN = (wave >> 2) * 64;  // 2 waves along N
  const int tileM = blockIdx.y * BM;
  const int tileN = blockIdx.x * BN;

  f32x8 acc[2][4];
  #pragma unroll
  for (int mt = 0; mt < 2; ++mt)
    #pragma unroll
    for (int nt = 0; nt < 4; ++nt)
      #pragma unroll
      for (int e = 0; e < 8; ++e) acc[mt][nt][e] = 0.f;

  const int mrow = lane & 15;
  const int akb  = (lane < 16) ? 0 : 8;    // A frag K-base (ISA 16-bit A layout)
  const int bkb  = (lane < 16) ? 0 : 8;    // B frag K-base (same K order as A)

  // this thread's two 16B chunks within a 128x32-half tile
  const int row0 = tid >> 2,          seg0 = tid & 3;
  const int row1 = (tid + 256) >> 2,  seg1 = (tid + 256) & 3;

  if (tileN + BN <= N){
    // ---------- fast path: async-to-LDS, double buffered ----------
    const u32 asb[2] = { (u32)(uintptr_t)&As[0][0], (u32)(uintptr_t)&As[1][0] };
    const u32 bsb[2] = { (u32)(uintptr_t)&Bs[0][0], (u32)(uintptr_t)&Bs[1][0] };

    auto issue = [&](int kt, int buf){
      async_ld_b128(asb[buf] + row0*(LDT*2) + seg0*16,
                    (u32)(((size_t)(tileM + row0)*K + kt + seg0*8) * 2), A);
      async_ld_b128(bsb[buf] + row0*(LDT*2) + seg0*16,
                    (u32)(((size_t)(tileN + row0)*K + kt + seg0*8) * 2), Bw);
      async_ld_b128(asb[buf] + row1*(LDT*2) + seg1*16,
                    (u32)(((size_t)(tileM + row1)*K + kt + seg1*8) * 2), A);
      async_ld_b128(bsb[buf] + row1*(LDT*2) + seg1*16,
                    (u32)(((size_t)(tileN + row1)*K + kt + seg1*8) * 2), Bw);
    };

    issue(0, 0);
    for (int kt = 0; kt < K; kt += BK){
      const int cb = (kt >> 5) & 1;
      if (kt + BK < K){
        issue(kt + BK, cb ^ 1);   // DMA next tile while we compute this one
        wait_async_le4();         // previous 4 ops (this tile) retired, next 4 in flight
      } else {
        wait_async_0();
      }
      __syncthreads();

      bf16x16 af[2], bfv[4];
      #pragma unroll
      for (int mt = 0; mt < 2; ++mt){
        const u16* p = &As[cb][0] + (waveM + mt*16 + mrow)*LDT + akb;
        af[mt] = frag_from(p, p + 16);
      }
      #pragma unroll
      for (int nt = 0; nt < 4; ++nt){
        const u16* p = &Bs[cb][0] + (waveN + nt*16 + mrow)*LDT + bkb;
        bfv[nt] = frag_from(p, p + 16);
      }
      #pragma unroll
      for (int mt = 0; mt < 2; ++mt)
        #pragma unroll
        for (int nt = 0; nt < 4; ++nt)
          acc[mt][nt] = __builtin_amdgcn_wmma_f32_16x16x32_bf16(
              false, af[mt], false, bfv[nt], (short)0, acc[mt][nt], false, false);
      __syncthreads();            // all waves done with buf cb before it is re-DMA'd
    }
  } else {
    // ---------- legacy path: guarded synchronous loads (partial N tiles) ----------
    for (int kt = 0; kt < K; kt += BK){
      #pragma unroll
      for (int it = 0; it < 2; ++it){
        int cc  = tid + it * 256;
        int row = cc >> 2, seg = cc & 3;
        *(uint4*)(&As[0][0] + row*LDT + seg*8) =
            *(const uint4*)(A + (size_t)(tileM + row)*K + kt + seg*8);
        uint4 bv; bv.x = bv.y = bv.z = bv.w = 0u;
        int gn = tileN + row;
        if (gn < N) bv = *(const uint4*)(Bw + (size_t)gn*K + kt + seg*8);
        *(uint4*)(&Bs[0][0] + row*LDT + seg*8) = bv;
      }
      if (kt + BK < K)
        __builtin_prefetch(A + (size_t)(tileM + (tid >> 1))*K + kt + BK, 0, 0);
      __syncthreads();

      bf16x16 af[2], bfv[4];
      #pragma unroll
      for (int mt = 0; mt < 2; ++mt){
        const u16* p = &As[0][0] + (waveM + mt*16 + mrow)*LDT + akb;
        af[mt] = frag_from(p, p + 16);
      }
      #pragma unroll
      for (int nt = 0; nt < 4; ++nt){
        const u16* p = &Bs[0][0] + (waveN + nt*16 + mrow)*LDT + bkb;
        bfv[nt] = frag_from(p, p + 16);
      }
      #pragma unroll
      for (int mt = 0; mt < 2; ++mt)
        #pragma unroll
        for (int nt = 0; nt < 4; ++nt)
          acc[mt][nt] = __builtin_amdgcn_wmma_f32_16x16x32_bf16(
              false, af[mt], false, bfv[nt], (short)0, acc[mt][nt], false, false);
      __syncthreads();
    }
  }

  // epilogue: D layout: VGPR e -> row (e + 8*(lane>=16)), col = lane&15
  const int rsel = (lane >> 4) << 3;
  const int ncol = lane & 15;
  #pragma unroll
  for (int mt = 0; mt < 2; ++mt){
    #pragma unroll
    for (int nt = 0; nt < 4; ++nt){
      int col = tileN + waveN + nt*16 + ncol;
      if (col < N){
        float cs = 1.f, bv = 0.f;
        if (logs){ float ls = logs[col / 768];
                   ls = fminf(fmaxf(ls, -10.f), 10.f);
                   cs = __expf(-ls); }            // (x/s)@W^T == (x@W^T)*exp(-clip(log_s))
        if (bias) bv = bias[col];
        #pragma unroll
        for (int e = 0; e < 8; ++e){
          int row = tileM + waveM + mt*16 + rsel + e;
          float v = acc[mt][nt][e] * cs + bv;
          if (relu) v = fmaxf(v, 0.f);
          size_t o = (size_t)row * N + col;
          if (res)  v += res[o];
          if (outF) outF[o] = v;
          if (outB) outB[o] = f2bf(v);
        }
      }
    }
  }
}

// ---------------- LayerNorm (optionally fused add), dual f32+bf16 output ----------------
__global__ __launch_bounds__(256)
void layernorm_kernel(const float* __restrict__ a, const float* __restrict__ b,
                      float* __restrict__ outF, u16* __restrict__ outB)
{
  const int C = NSTATE;
  const int row = blockIdx.x;
  const int t = threadIdx.x;
  __shared__ float ps[8], ps2[8];
  float x[4]; float s = 0.f, s2 = 0.f;
  #pragma unroll
  for (int i = 0; i < 4; ++i){
    int c = t + i * 256;
    float v = a[(size_t)row*C + c];
    if (b) v += b[(size_t)row*C + c];
    x[i] = v; s += v; s2 += v*v;
  }
  #pragma unroll
  for (int o = 16; o > 0; o >>= 1){ s += __shfl_xor(s, o, 32); s2 += __shfl_xor(s2, o, 32); }
  if ((t & 31) == 0){ ps[t >> 5] = s; ps2[t >> 5] = s2; }
  __syncthreads();
  float S = 0.f, S2 = 0.f;
  #pragma unroll
  for (int i = 0; i < 8; ++i){ S += ps[i]; S2 += ps2[i]; }
  float mean = S * (1.f / C);
  float var  = S2 * (1.f / C) - mean * mean;
  float rstd = rsqrtf(var + 1e-5f);
  #pragma unroll
  for (int i = 0; i < 4; ++i){
    int c = t + i * 256;
    float v = (x[i] - mean) * rstd;
    size_t o = (size_t)row*C + c;
    if (outF) outF[o] = v;
    if (outB) outB[o] = f2bf(v);
  }
}

// ---------------- persistent GRU scan, one workgroup per channel ----------------
// LDS: whh r,z rows (512 x 258 bf16, padded -> bank-conflict-free) + h pingpong.
#define SCAN_LDS (512*258*2 + 2*GH*4)

__global__ __launch_bounds__(256, 1)
void scan_kernel(const float* __restrict__ ig,   // [T, IGW], channel slice at ch*768
                 const u16*  __restrict__ whh,   // bf16 [NCH][768][GH] (layer base)
                 const float* __restrict__ bn,   // [NCH][GH]
                 const float* __restrict__ logs, // [NCH]
                 const float* __restrict__ h0,   // [GH]
                 float* __restrict__ y)          // [T, NSTATE]
{
  extern __shared__ char smem[];
  u16*   wrz = (u16*)smem;                    // [512][258]
  float* hb  = (float*)(smem + 512*258*2);    // [2][GH]
  const int u  = threadIdx.x;                 // hidden unit owned by this thread
  const int ch = blockIdx.x;
  const u16* wch = whh + (size_t)ch * 768 * GH;

  // stage r,z weight rows into LDS (u32 granularity; row stride 258 halves)
  for (int idx = u; idx < 512 * 128; idx += 256){
    int r = idx >> 7, k2 = idx & 127;
    *(u32*)((char*)wrz + 516*r + 4*k2) = *(const u32*)(wch + r*GH + k2*2);
  }
  // n-row of this unit lives in VGPRs (128 dwords of packed bf16)
  u32 wn[128];
  {
    const uint4* p = (const uint4*)(wch + (size_t)(512 + u) * GH);
    #pragma unroll
    for (int i = 0; i < 32; ++i){
      uint4 q = p[i];
      wn[4*i] = q.x; wn[4*i+1] = q.y; wn[4*i+2] = q.z; wn[4*i+3] = q.w;
    }
  }

  float ls = logs[ch]; ls = fminf(fmaxf(ls, -10.f), 10.f);
  float inv_s = 1.f / __expf(ls);
  float bnu = bn[ch*GH + u];
  float hu  = h0[u];
  hb[u] = hu;
  __syncthreads();

  const float* igc = ig + ch * (3 * GH);
  float* yc = y + ch * GH;
  const char* wr_row = (const char*)wrz + 516 * u;
  const char* wz_row = (const char*)wrz + 516 * (256 + u);

  for (int t = 0; t < T_STEPS; ++t){
    const float* hcur = hb + (t & 1) * GH;
    float accr = 0.f, accz = 0.f, accn = 0.f;
    #pragma unroll
    for (int i = 0; i < 128; ++i){
      float2 h2 = *(const float2*)(hcur + 2*i);   // LDS broadcast read
      u32 wr  = *(const u32*)(wr_row + 4*i);
      u32 wz  = *(const u32*)(wz_row + 4*i);
      u32 wnn = wn[i];
      accr = fmaf(bf_lo(wr),  h2.x, accr);  accr = fmaf(bf_hi(wr),  h2.y, accr);
      accz = fmaf(bf_lo(wz),  h2.x, accz);  accz = fmaf(bf_hi(wz),  h2.y, accz);
      accn = fmaf(bf_lo(wnn), h2.x, accn);  accn = fmaf(bf_hi(wnn), h2.y, accn);
    }
    const float* igt = igc + (size_t)t * IGW;
    float r = 1.f / (1.f + __expf(-(igt[u]        + accr)));
    float z = 1.f / (1.f + __expf(-(igt[GH + u]   + accz)));
    float n = tanhf(igt[2*GH + u] + r * (accn + bnu));
    float hnew = n + z * (hu - n);
    float hout = (hnew - hu) * inv_s + hu;      // ScaleGRU rescale
    yc[(size_t)t * NSTATE + u] = hout;
    hb[((t + 1) & 1) * GH + u] = hout;          // pingpong: single barrier/step
    hu = hout;
    __syncthreads();
  }
}

// ---------------- host orchestration ----------------
extern "C" void kernel_launch(void* const* d_in, const int* in_sizes, int n_in,
                              void* d_out, int out_size, void* d_ws, size_t ws_size,
                              hipStream_t stream)
{
  (void)in_sizes; (void)n_in; (void)out_size; (void)ws_size;
  const float* inputs   = (const float*)d_in[0];
  const float* h0       = (const float*)d_in[1];
  // d_in[2] yinit_guess: unused (exact fixed point is independent of it)
  const float* enc_w1   = (const float*)d_in[3];
  const float* enc_b1   = (const float*)d_in[4];
  const float* enc_w2   = (const float*)d_in[5];
  const float* enc_b2   = (const float*)d_in[6];
  const float* gru_wih  = (const float*)d_in[7];
  const float* gru_whh  = (const float*)d_in[8];
  const float* gru_b    = (const float*)d_in[9];
  const float* gru_bn   = (const float*)d_in[10];
  const float* gru_logs = (const float*)d_in[11];
  const float* mlp_w1   = (const float*)d_in[12];
  const float* mlp_b1   = (const float*)d_in[13];
  const float* mlp_w2   = (const float*)d_in[14];
  const float* mlp_b2   = (const float*)d_in[15];
  const float* cls_w1   = (const float*)d_in[16];
  const float* cls_b1   = (const float*)d_in[17];
  const float* cls_w2   = (const float*)d_in[18];
  const float* cls_b2   = (const float*)d_in[19];
  float* out = (float*)d_out;

  char* base = (char*)d_ws;
  size_t off = 0;
  auto alloc = [&](size_t bytes) -> char* {
    char* p = base + off;
    off += (bytes + 255) & ~(size_t)255;
    return p;
  };
  // workspace (~345 MB total)
  u16*   inp_b = (u16*)  alloc((size_t)T_STEPS*NINP*2);
  u16*   ew1b  = (u16*)  alloc((size_t)NSTATE*NINP*2);
  u16*   ew2b  = (u16*)  alloc((size_t)NSTATE*NSTATE*2);
  u16*   wihb  = (u16*)  alloc((size_t)NLAYER*NCH*768*NSTATE*2);
  u16*   whhb  = (u16*)  alloc((size_t)NLAYER*NCH*768*GH*2);
  u16*   m1b   = (u16*)  alloc((size_t)NLAYER*NSTATE*NSTATE*2);
  u16*   m2b   = (u16*)  alloc((size_t)NLAYER*NSTATE*NSTATE*2);
  u16*   c1b   = (u16*)  alloc((size_t)NSTATE*NSTATE*2);
  u16*   c2b   = (u16*)  alloc((size_t)NCLASS*NSTATE*2);
  u16*   Hb    = (u16*)  alloc((size_t)T_STEPS*NSTATE*2);
  float* Xf    = (float*)alloc((size_t)T_STEPS*NSTATE*4);
  u16*   Xb    = (u16*)  alloc((size_t)T_STEPS*NSTATE*2);
  float* XIf   = (float*)alloc((size_t)T_STEPS*NSTATE*4);
  u16*   XIb   = (u16*)  alloc((size_t)T_STEPS*NSTATE*2);
  float* Yf    = (float*)alloc((size_t)T_STEPS*NSTATE*4);
  float* Y2f   = (float*)alloc((size_t)T_STEPS*NSTATE*4);
  u16*   Y2b   = (u16*)  alloc((size_t)T_STEPS*NSTATE*2);
  float* IG    = (float*)alloc((size_t)T_STEPS*IGW*4);

  auto cvt = [&](const float* src, u16* dst, long n){
    int g = (int)((n + 1023) / 1024); if (g > 8192) g = 8192; if (g < 1) g = 1;
    cvt_f32_bf16<<<g, 256, 0, stream>>>(src, dst, n);
  };
  cvt(inputs,  inp_b, (long)T_STEPS*NINP);
  cvt(enc_w1,  ew1b,  (long)NSTATE*NINP);
  cvt(enc_w2,  ew2b,  (long)NSTATE*NSTATE);
  cvt(gru_wih, wihb,  (long)NLAYER*NCH*768*NSTATE);
  cvt(gru_whh, whhb,  (long)NLAYER*NCH*768*GH);
  cvt(mlp_w1,  m1b,   (long)NLAYER*NSTATE*NSTATE);
  cvt(mlp_w2,  m2b,   (long)NLAYER*NSTATE*NSTATE);
  cvt(cls_w1,  c1b,   (long)NSTATE*NSTATE);
  cvt(cls_w2,  c2b,   (long)NCLASS*NSTATE);

  auto gemm = [&](const u16* A, const u16* Bw, const float* bias, const float* logs,
                  const float* res, float* oF, u16* oB, int M, int N, int K, int relu){
    dim3 g((N + BN - 1) / BN, M / BM);
    gemm_bf16_wmma<<<g, 256, 0, stream>>>(A, Bw, bias, logs, res, oF, oB, M, N, K, relu);
  };

  // encoder: relu(x@W1^T+b1)@W2^T+b2
  gemm(inp_b, ew1b, enc_b1, nullptr, nullptr, nullptr, Hb, T_STEPS, NSTATE, NINP, 1);
  gemm(Hb,    ew2b, enc_b2, nullptr, nullptr, Xf, nullptr, T_STEPS, NSTATE, NSTATE, 0);

  (void)hipFuncSetAttribute((const void*)scan_kernel,
                            hipFuncAttributeMaxDynamicSharedMemorySize, SCAN_LDS);

  for (int l = 0; l < NLAYER; ++l){
    layernorm_kernel<<<T_STEPS, 256, 0, stream>>>(Xf, nullptr, XIf, XIb);
    // ig for all 4 channels in one GEMM: [8192,1024] x [3072,1024]^T, scale+bias fused
    gemm(XIb, wihb + (size_t)l*NCH*768*NSTATE, gru_b + (size_t)l*IGW,
         gru_logs + (size_t)l*NCH, nullptr, IG, nullptr, T_STEPS, IGW, NSTATE, 0);
    scan_kernel<<<NCH, GH, SCAN_LDS, stream>>>(
         IG, whhb + (size_t)l*NCH*768*GH, gru_bn + (size_t)l*NCH*GH,
         gru_logs + (size_t)l*NCH, h0, Yf);
    layernorm_kernel<<<T_STEPS, 256, 0, stream>>>(Yf, XIf, Y2f, Y2b);   // LN(y+xi)
    gemm(Y2b, m1b + (size_t)l*NSTATE*NSTATE, mlp_b1 + (size_t)l*NSTATE,
         nullptr, nullptr, nullptr, Hb, T_STEPS, NSTATE, NSTATE, 1);
    gemm(Hb,  m2b + (size_t)l*NSTATE*NSTATE, mlp_b2 + (size_t)l*NSTATE,
         nullptr, Y2f, Xf, Xb, T_STEPS, NSTATE, NSTATE, 0);             // +residual
  }

  // classifier
  gemm(Xb, c1b, cls_b1, nullptr, nullptr, nullptr, Hb, T_STEPS, NSTATE, NSTATE, 1);
  gemm(Hb, c2b, cls_b2, nullptr, nullptr, out, nullptr, T_STEPS, NCLASS, NSTATE, 0);
}
